// RouteNet_81346680586710
// MI455X (gfx1250) — matrix operand order repaired and checked by reading, not compiled, not run on hardware
//
#include <hip/hip_runtime.h>
#include <hip/hip_bf16.h>
#include <stdint.h>

#define N_NODES 50000
#define N_EDGES 400000
#define E_TOT   450000     // edges + self loops
#define F1 512             // layer-1 feature width (4 heads * 128)
#define F2 256             // layer-2 feature width (2 heads * 128)
#define HID 128
#define H1 4
#define H2 2
#define NEG_SLOPE 0.2f

typedef __attribute__((ext_vector_type(16))) __bf16 v16bf;
typedef __attribute__((ext_vector_type(8)))  float  v8f;

__device__ __forceinline__ float leaky(float v) { return v > 0.f ? v : NEG_SLOPE * v; }

// order-preserving float <-> uint encoding for atomic max (all real keys > 0)
__device__ __forceinline__ unsigned fenc(float f) {
    unsigned u = __float_as_uint(f);
    return (u & 0x80000000u) ? ~u : (u | 0x80000000u);
}
__device__ __forceinline__ float fdec(unsigned k) {
    return __uint_as_float((k & 0x80000000u) ? (k & 0x7FFFFFFFu) : ~k);
}
__device__ __forceinline__ uint16_t f2bf(float f) {
    unsigned u = __float_as_uint(f);
    u += 0x7FFFu + ((u >> 16) & 1u);          // round-to-nearest-even
    return (uint16_t)(u >> 16);
}
__device__ __forceinline__ void edge_sd(const int* __restrict__ ei, int e, int& s, int& d) {
    if (e < N_EDGES) { s = ei[e]; d = ei[N_EDGES + e]; }
    else             { s = d = e - N_EDGES; }   // self loop
}

// ---------- layer-1 linear: hlin1[m,n] = sum_k x[m,k]*W1[k,n], K=3 ----------
__global__ void k_gemm1(const float* __restrict__ x, const float* __restrict__ W1,
                        float* __restrict__ hlin1) {
    int idx = blockIdx.x * blockDim.x + threadIdx.x;
    if (idx >= N_NODES * F1) return;
    int m = idx >> 9, n = idx & (F1 - 1);
    const float* xr = x + m * 3;
    hlin1[idx] = fmaf(xr[0], W1[n], fmaf(xr[1], W1[F1 + n], xr[2] * W1[2 * F1 + n]));
}

// ---------- per-node attention coefficients es/ed ----------
template <int H, int F>
__global__ void k_esed(const float* __restrict__ h, const float* __restrict__ asrc,
                       const float* __restrict__ adst, float* __restrict__ es,
                       float* __restrict__ ed) {
    int idx = blockIdx.x * blockDim.x + threadIdx.x;
    if (idx >= N_NODES * H) return;
    int node = idx / H, hh = idx % H;
    const float* hr = h + node * F + hh * HID;
    const float* as = asrc + hh * HID;
    const float* ad = adst + hh * HID;
    float s = 0.f, d = 0.f;
#pragma unroll 4
    for (int c = 0; c < HID; ++c) { float v = hr[c]; s = fmaf(v, as[c], s); d = fmaf(v, ad[c], d); }
    es[idx] = s; ed[idx] = d;
}

// ---------- segment max over incoming edges (encoded-uint atomic max) ----------
template <int H>
__global__ void k_edge_max(const int* __restrict__ ei, const float* __restrict__ es,
                           const float* __restrict__ ed, unsigned* __restrict__ mkey) {
    int e = blockIdx.x * blockDim.x + threadIdx.x;
    if (e >= E_TOT) return;
    int s, d; edge_sd(ei, e, s, d);
#pragma unroll
    for (int h = 0; h < H; ++h) {
        float v = leaky(es[s * H + h] + ed[d * H + h]);
        atomicMax(&mkey[d * H + h], fenc(v));
    }
}

// ---------- exp(e - max) and segment-sum denominator ----------
template <int H>
__global__ void k_edge_exp(const int* __restrict__ ei, const float* __restrict__ es,
                           const float* __restrict__ ed, const unsigned* __restrict__ mkey,
                           float* __restrict__ exbuf, float* __restrict__ denom) {
    int e = blockIdx.x * blockDim.x + threadIdx.x;
    if (e >= E_TOT) return;
    int s, d; edge_sd(ei, e, s, d);
#pragma unroll
    for (int h = 0; h < H; ++h) {
        float v  = leaky(es[s * H + h] + ed[d * H + h]);
        float ex = expf(v - fdec(mkey[d * H + h]));
        exbuf[e * H + h] = ex;
        atomicAdd(&denom[d * H + h], ex);
    }
}

// ---------- weighted scatter-add aggregation (one block of 256 per edge) ----------
template <int H, int F>
__global__ void k_agg(const int* __restrict__ ei, const float* __restrict__ hlin,
                      const float* __restrict__ exbuf, const float* __restrict__ denom,
                      float* __restrict__ agg) {
    int e = blockIdx.x, t = threadIdx.x;
    int s, d; edge_sd(ei, e, s, d);
    int c = t & (HID - 1);
    for (int h = t >> 7; h < H; h += 2) {                    // 256 threads -> 2 head-slices
        float alpha = exbuf[e * H + h] / (denom[d * H + h] + 1e-16f);
        atomicAdd(&agg[d * F + h * HID + c], hlin[s * F + h * HID + c] * alpha);
    }
}

// ---------- layer-1 epilogue: +b1, ReLU, convert to bf16 for WMMA GEMM ----------
__global__ void k_epi1(const float* __restrict__ agg, const float* __restrict__ b1,
                       uint16_t* __restrict__ h1bf) {
    int idx = blockIdx.x * blockDim.x + threadIdx.x;
    if (idx >= N_NODES * F1) return;
    int n = idx & (F1 - 1);
    float v = agg[idx] + b1[n];
    h1bf[idx] = f2bf(v > 0.f ? v : 0.f);
}

// ---------- W2 [512,256] f32 -> W2T [256,512] bf16 (K-contiguous per column) ----------
__global__ void k_w2t(const float* __restrict__ W2, uint16_t* __restrict__ W2T) {
    int idx = blockIdx.x * blockDim.x + threadIdx.x;
    if (idx >= F1 * F2) return;
    int k = idx >> 8, n = idx & (F2 - 1);
    W2T[n * F1 + k] = f2bf(W2[idx]);
}

// ---------- layer-2 linear via v_wmma_f32_16x16x32_bf16: one wave per 16x16 tile ----------
__global__ void k_gemm2(const uint16_t* __restrict__ Abf,   // [N,512] bf16
                        const uint16_t* __restrict__ BT,    // [256,512] bf16 (K-contig)
                        float* __restrict__ C) {            // [N,256] f32
    int wave = blockIdx.x * (blockDim.x >> 5) + (threadIdx.x >> 5);  // 50000 tiles
    int tM = wave >> 4;                 // 0..3124
    int tN = wave & 15;                 // 0..15
    int lane = threadIdx.x & 31;
    int half = lane >> 4, r = lane & 15;

    const uint16_t* ap = Abf + (tM * 16 + r) * F1;            // A row, per-lane
    const uint16_t* bp = BT + (tN * 16 + r) * F1 + half * 16; // B column, K-half by lane-half
    v8f acc = {};
    for (int k0 = 0; k0 < F1; k0 += 32) {
        union { v16bf v; uint4 u[2]; } a, b;
        // A k order (contiguous, matches B): lane-half 0 -> K k0..k0+15; half 1 -> k0+16..k0+31
        a.u[0] = *(const uint4*)(ap + k0 + half * 16);
        a.u[1] = *(const uint4*)(ap + k0 + half * 16 + 8);
        // B k order (contiguous, matches A): lane-half 0 -> K k0..k0+15; half 1 -> k0+16..k0+31
        b.u[0] = *(const uint4*)(bp + k0);
        b.u[1] = *(const uint4*)(bp + k0 + 8);
        acc = __builtin_amdgcn_wmma_f32_16x16x32_bf16(false, a.v, false, b.v,
                                                      (short)0, acc, false, false);
    }
    // D layout: VGPR j, lane-half 0 -> M=j, half 1 -> M=8+j; N = r
    float* cp = C + (tM * 16 + half * 8) * F2 + tN * 16 + r;
#pragma unroll
    for (int j = 0; j < 8; ++j) cp[j * F2] = acc[j];
}

// ---------- layer-2 epilogue: mean over 2 heads, +b2, ReLU ----------
__global__ void k_epi2(const float* __restrict__ agg2, const float* __restrict__ b2,
                       float* __restrict__ hfinal) {
    int idx = blockIdx.x * blockDim.x + threadIdx.x;
    if (idx >= N_NODES * HID) return;
    int m = idx >> 7, c = idx & (HID - 1);
    float v = 0.5f * (agg2[m * F2 + c] + agg2[m * F2 + HID + c]) + b2[c];
    hfinal[idx] = v > 0.f ? v : 0.f;
}

// ---------- final scorer ----------
__global__ void k_score(const float* __restrict__ hfinal, const float* __restrict__ fcw,
                        const float* __restrict__ fcb, float* __restrict__ scores) {
    int m = blockIdx.x * blockDim.x + threadIdx.x;
    if (m >= N_NODES) return;
    const float* hr = hfinal + m * HID;
    float acc = 0.f;
#pragma unroll 4
    for (int c = 0; c < HID; ++c) acc = fmaf(hr[c], fcw[c], acc);
    scores[m] = acc + fcb[0];
}

// ---------- global softmax over 50000 scores ----------
__global__ void k_rmax(const float* __restrict__ scores, unsigned* __restrict__ rmax) {
    __shared__ float sm[256];
    int i = blockIdx.x * blockDim.x + threadIdx.x;
    sm[threadIdx.x] = (i < N_NODES) ? scores[i] : -3.4e38f;
    __syncthreads();
    for (int o = 128; o > 0; o >>= 1) {
        if (threadIdx.x < o) sm[threadIdx.x] = fmaxf(sm[threadIdx.x], sm[threadIdx.x + o]);
        __syncthreads();
    }
    if (threadIdx.x == 0) atomicMax(rmax, fenc(sm[0]));
}
__global__ void k_rsum(const float* __restrict__ scores, const unsigned* __restrict__ rmax,
                       float* __restrict__ rsum) {
    __shared__ float sm[256];
    float mx = fdec(*rmax);
    int i = blockIdx.x * blockDim.x + threadIdx.x;
    sm[threadIdx.x] = (i < N_NODES) ? expf(scores[i] - mx) : 0.f;
    __syncthreads();
    for (int o = 128; o > 0; o >>= 1) {
        if (threadIdx.x < o) sm[threadIdx.x] += sm[threadIdx.x + o];
        __syncthreads();
    }
    if (threadIdx.x == 0) atomicAdd(rsum, sm[0]);
}
__global__ void k_final(const float* __restrict__ scores, const unsigned* __restrict__ rmax,
                        const float* __restrict__ rsum, float* __restrict__ out) {
    int i = blockIdx.x * blockDim.x + threadIdx.x;
    if (i >= N_NODES) return;
    out[i] = expf(scores[i] - fdec(*rmax)) / (*rsum);
}

extern "C" void kernel_launch(void* const* d_in, const int* in_sizes, int n_in,
                              void* d_out, int out_size, void* d_ws, size_t ws_size,
                              hipStream_t stream) {
    const float* x   = (const float*)d_in[0];
    const int*   ei  = (const int*)d_in[1];
    const float* W1  = (const float*)d_in[2];
    const float* as1 = (const float*)d_in[3];
    const float* ad1 = (const float*)d_in[4];
    const float* b1  = (const float*)d_in[5];
    const float* W2  = (const float*)d_in[6];
    const float* as2 = (const float*)d_in[7];
    const float* ad2 = (const float*)d_in[8];
    const float* b2  = (const float*)d_in[9];
    const float* fcw = (const float*)d_in[10];
    const float* fcb = (const float*)d_in[11];
    float* out = (float*)d_out;

    // ---- workspace layout with reuse (~269 MB) ----
    char*  ws  = (char*)d_ws;
    size_t off = 0;
    auto carve = [&](size_t bytes) -> char* {
        char* p = ws + off;
        off += (bytes + 255) & ~(size_t)255;
        return p;
    };
    float*    hlin1  = (float*)carve((size_t)N_NODES * F1 * 4);  // reused as hlin2
    float*    agg1   = (float*)carve((size_t)N_NODES * F1 * 4);  // reused as agg2
    uint16_t* h1bf   = (uint16_t*)carve((size_t)N_NODES * F1 * 2); // reused as hfinal(f32)
    uint16_t* w2t    = (uint16_t*)carve((size_t)F1 * F2 * 2);
    float*    es1    = (float*)carve((size_t)N_NODES * H1 * 4);
    float*    ed1    = (float*)carve((size_t)N_NODES * H1 * 4);
    unsigned* m1     = (unsigned*)carve((size_t)N_NODES * H1 * 4);
    float*    den1   = (float*)carve((size_t)N_NODES * H1 * 4);
    float*    es2    = (float*)carve((size_t)N_NODES * H2 * 4);
    float*    ed2    = (float*)carve((size_t)N_NODES * H2 * 4);
    unsigned* m2     = (unsigned*)carve((size_t)N_NODES * H2 * 4);
    float*    den2   = (float*)carve((size_t)N_NODES * H2 * 4);
    float*    exbuf  = (float*)carve((size_t)E_TOT * H1 * 4);    // reused for layer 2
    float*    scores = (float*)carve((size_t)N_NODES * 4);
    unsigned* red    = (unsigned*)carve(256);                    // [0]=max key, [1]=sum bits

    float* hlin2  = hlin1;
    float* agg2   = agg1;
    float* hfinal = (float*)h1bf;
    float* exbuf2 = exbuf;

    const int B = 256;
    // scratch init (zero is valid for encoded-max keys: every real key > 0)
    hipMemsetAsync(m1,   0, (size_t)N_NODES * H1 * 4, stream);
    hipMemsetAsync(den1, 0, (size_t)N_NODES * H1 * 4, stream);
    hipMemsetAsync(agg1, 0, (size_t)N_NODES * F1 * 4, stream);
    hipMemsetAsync(m2,   0, (size_t)N_NODES * H2 * 4, stream);
    hipMemsetAsync(den2, 0, (size_t)N_NODES * H2 * 4, stream);
    hipMemsetAsync(red,  0, 256, stream);

    // ---- layer 1 ----
    k_gemm1<<<(N_NODES * F1 + B - 1) / B, B, 0, stream>>>(x, W1, hlin1);
    k_esed<H1, F1><<<(N_NODES * H1 + B - 1) / B, B, 0, stream>>>(hlin1, as1, ad1, es1, ed1);
    k_edge_max<H1><<<(E_TOT + B - 1) / B, B, 0, stream>>>(ei, es1, ed1, m1);
    k_edge_exp<H1><<<(E_TOT + B - 1) / B, B, 0, stream>>>(ei, es1, ed1, m1, exbuf, den1);
    k_agg<H1, F1><<<E_TOT, B, 0, stream>>>(ei, hlin1, exbuf, den1, agg1);
    k_epi1<<<(N_NODES * F1 + B - 1) / B, B, 0, stream>>>(agg1, b1, h1bf);

    // ---- layer 2 ----
    hipMemsetAsync(agg2, 0, (size_t)N_NODES * F2 * 4, stream);   // after agg1 consumed
    k_w2t<<<(F1 * F2 + B - 1) / B, B, 0, stream>>>(W2, w2t);
    k_gemm2<<<(N_NODES / 16) * (F2 / 16) / 8, B, 0, stream>>>(h1bf, w2t, hlin2); // 6250 blocks
    k_esed<H2, F2><<<(N_NODES * H2 + B - 1) / B, B, 0, stream>>>(hlin2, as2, ad2, es2, ed2);
    k_edge_max<H2><<<(E_TOT + B - 1) / B, B, 0, stream>>>(ei, es2, ed2, m2);
    k_edge_exp<H2><<<(E_TOT + B - 1) / B, B, 0, stream>>>(ei, es2, ed2, m2, exbuf2, den2);
    k_agg<H2, F2><<<E_TOT, B, 0, stream>>>(ei, hlin2, exbuf2, den2, agg2);
    k_epi2<<<(N_NODES * HID + B - 1) / B, B, 0, stream>>>(agg2, b2, hfinal);

    // ---- scorer + global softmax ----
    k_score<<<(N_NODES + B - 1) / B, B, 0, stream>>>(hfinal, fcw, fcb, scores);
    k_rmax<<<(N_NODES + B - 1) / B, B, 0, stream>>>(scores, &red[0]);
    k_rsum<<<(N_NODES + B - 1) / B, B, 0, stream>>>(scores, &red[0], (float*)&red[1]);
    k_final<<<(N_NODES + B - 1) / B, B, 0, stream>>>(scores, &red[0], (const float*)&red[1], out);
}
